// PointNetSetAbstractionMsg_28286654611732
// MI455X (gfx1250) — compile-verified, matrix operand order checked by reading, not run on hardware
//
#include <hip/hip_runtime.h>

typedef _Float16 h16;
typedef __attribute__((ext_vector_type(16))) _Float16 v16h;
typedef __attribute__((ext_vector_type(8)))  float    v8f;

#define B_  16
#define N_  4096
#define S_  512
#define BN_EPS 1e-5f

// ---- CDNA5 16-bit A/B fragment lane layout (CDNA5 ISA 7.12.2) ----
// lane holds m (or n) = lane&15; half h of the v16h covers
// k = 16*(h>>3) + 8*(lane>>4) + 2*((h>>1)&3) + (h&1)
__device__ __forceinline__ int frag_k(int lane, int h) {
  return ((h >> 3) << 4) | ((lane >> 4) << 3) | (((h >> 1) & 3) << 1) | (h & 1);
}

// =====================  1. Farthest point sampling  =====================
__global__ void fps_kernel(const float* __restrict__ xyz, int* __restrict__ fps_idx,
                           float* __restrict__ nxyz_ws /*(B,S,3)*/,
                           float* __restrict__ out /* d_out, (B,3,S) part */) {
  __shared__ float dist[N_];
  __shared__ float rv[256];
  __shared__ int   ri[256];
  __shared__ int   cur;
  const int b = blockIdx.x, t = threadIdx.x;
  const float* xb = xyz + (size_t)b * 3 * N_;
  for (int i = t; i < N_; i += 256) dist[i] = 1e10f;
  if (t == 0) cur = 0;
  __syncthreads();
  for (int it = 0; it < S_; ++it) {
    const int f = cur;
    const float cx = xb[f], cy = xb[N_ + f], cz = xb[2 * N_ + f];
    if (t == 0) {
      fps_idx[b * S_ + it] = f;
      float* nw = nxyz_ws + ((size_t)b * S_ + it) * 3;
      nw[0] = cx; nw[1] = cy; nw[2] = cz;
      out[(size_t)b * 3 * S_ + 0 * S_ + it] = cx;
      out[(size_t)b * 3 * S_ + 1 * S_ + it] = cy;
      out[(size_t)b * 3 * S_ + 2 * S_ + it] = cz;
    }
    float bv = -1.0f; int bi = 0;
    for (int i = t; i < N_; i += 256) {
      const float dx = xb[i] - cx, dy = xb[N_ + i] - cy, dz = xb[2 * N_ + i] - cz;
      const float nd = fminf(dist[i], dx * dx + dy * dy + dz * dz);
      dist[i] = nd;
      if (nd > bv) { bv = nd; bi = i; }
    }
    rv[t] = bv; ri[t] = bi;
    __syncthreads();
    for (int off = 128; off > 0; off >>= 1) {
      if (t < off) {
        if (rv[t + off] > rv[t] || (rv[t + off] == rv[t] && ri[t + off] < ri[t])) {
          rv[t] = rv[t + off]; ri[t] = ri[t + off];
        }
      }
      __syncthreads();
    }
    if (t == 0) cur = ri[0];
    __syncthreads();
  }
}

// =====================  2. Ball query (wave32 stream compaction)  =====================
__global__ void ballquery_kernel(const float* __restrict__ xyz, const float* __restrict__ nxyz,
                                 int* __restrict__ gidx, float r2, int K) {
  const int wid  = (blockIdx.x * blockDim.x + threadIdx.x) >> 5;
  const int lane = threadIdx.x & 31;
  if (wid >= B_ * S_) return;
  const int b = wid / S_;
  const float* xb = xyz + (size_t)b * 3 * N_;
  const float* c  = nxyz + (size_t)wid * 3;
  const float cx = c[0], cy = c[1], cz = c[2];
  int* g = gidx + (size_t)wid * K;
  int cnt = 0, firstn = 0;
  for (int n0 = 0; n0 < N_; n0 += 32) {
    const int n = n0 + lane;
    const float dx = xb[n] - cx, dy = xb[N_ + n] - cy, dz = xb[2 * N_ + n] - cz;
    const bool ok = (dx * dx + dy * dy + dz * dz) <= r2;
    const unsigned m = (unsigned)__ballot(ok);
    if (cnt == 0 && m != 0) firstn = n0 + (__ffs((int)m) - 1);
    const int pos = cnt + __popc(m & ((1u << lane) - 1u));
    if (ok && pos < K) g[pos] = n;
    cnt += __popc(m);
    if (cnt >= K) break;   // cnt is wave-uniform
  }
  for (int p = cnt + lane; p < K; p += 32) g[p] = firstn;  // pad with first hit
}

// =====================  shared WMMA epilogue: bias, store fp16, BN stats =====================
__device__ __forceinline__ void epilogue(v8f acc, const float* __restrict__ bias,
                                         h16* __restrict__ X, float* __restrict__ sum,
                                         float* __restrict__ sumsq, int r0, int ct, int co, int lane) {
  const int nn = lane & 15, o = ct * 16 + nn, mb = (lane >> 4) << 3;
  const float bv = bias[o];
  float s = 0.f, ss = 0.f;
  h16* Xp = X + (size_t)(r0 + mb) * co + o;
#pragma unroll
  for (int i = 0; i < 8; ++i) {                 // C/D layout: VGPR i -> (M=mb+i, N=nn)
    const float v = acc[i] + bv;
    Xp[(size_t)i * co] = (h16)v;
    s += v; ss += v * v;
  }
  s  += __shfl_xor(s, 16);                      // combine M=0..7 and M=8..15 halves
  ss += __shfl_xor(ss, 16);
  if (lane < 16) { atomicAdd(sum + o, s); atomicAdd(sumsq + o, ss); }
}

// =====================  3a. Layer-1: gather + GEMM (cin=6 padded to 32)  =====================
template <int CO>
__global__ void gemm_l1_kernel(const float* __restrict__ xyz, const float* __restrict__ pts,
                               const float* __restrict__ nxyz, const int* __restrict__ gidx,
                               const float* __restrict__ W, const float* __restrict__ bias,
                               h16* __restrict__ X, float* __restrict__ sum,
                               float* __restrict__ sumsq, int K) {
  __shared__ h16 sW[CO * 32];
  __shared__ h16 sA[8][16 * 32];
  const int t = threadIdx.x, wv = t >> 5, lane = t & 31;
  for (int i = t; i < CO * 32; i += 256) {      // stage W zero-padded to cin=32
    const int o = i >> 5, k = i & 31;
    sW[i] = (k < 6) ? (h16)W[o * 6 + k] : (h16)0.f;
  }
  __syncthreads();
  const int ntiles = (B_ * S_ * K) >> 4;
  const int wstride = gridDim.x * 8;
  h16* Aw = sA[wv];
  for (int tile = blockIdx.x * 8 + wv; tile < ntiles; tile += wstride) {
    const int r0 = tile << 4;
    if (lane < 16) {                            // gather one row: [pts(3), xyz-center(3), 0...]
      const int r = r0 + lane;
      const int b = r / (S_ * K), rem = r % (S_ * K), s = rem / K, kk = rem % K;
      const int n = gidx[((size_t)b * S_ + s) * (size_t)K + kk];
      const float* xb = xyz + (size_t)b * 3 * N_;
      const float* pb = pts + (size_t)b * 3 * N_;
      const float* c  = nxyz + ((size_t)b * S_ + s) * 3;
      h16* row = Aw + lane * 32;
      row[0] = (h16)pb[n]; row[1] = (h16)pb[N_ + n]; row[2] = (h16)pb[2 * N_ + n];
      row[3] = (h16)(xb[n] - c[0]); row[4] = (h16)(xb[N_ + n] - c[1]); row[5] = (h16)(xb[2 * N_ + n] - c[2]);
#pragma unroll
      for (int j = 6; j < 32; ++j) row[j] = (h16)0.f;
    }
    asm volatile("s_wait_dscnt 0" ::: "memory");  // cross-lane LDS store->load in same wave
    v16h a;
    const int m = lane & 15, nn = lane & 15;
#pragma unroll
    for (int h = 0; h < 16; ++h) a[h] = Aw[m * 32 + frag_k(lane, h)];
#pragma unroll
    for (int ct = 0; ct < (CO >> 4); ++ct) {
      v16h bf;
#pragma unroll
      for (int h = 0; h < 16; ++h) bf[h] = sW[(ct * 16 + nn) * 32 + frag_k(lane, h)];
      v8f acc = {};
      acc = __builtin_amdgcn_wmma_f32_16x16x32_f16(false, a, false, bf, (short)0, acc, false, false);
      epilogue(acc, bias, X, sum, sumsq, r0, ct, CO, lane);
    }
  }
}

// =====================  3b. Layers 2/3: BN+ReLU fused into A staging, WMMA GEMM =====================
template <int CIN, int CO>
__global__ void gemm_bn_kernel(const h16* __restrict__ A, const float* __restrict__ scale,
                               const float* __restrict__ shift, const float* __restrict__ W,
                               const float* __restrict__ bias, h16* __restrict__ X,
                               float* __restrict__ sum, float* __restrict__ sumsq, int R) {
  __shared__ h16 sW[CO * CIN];
  __shared__ h16 sA[8][16 * CIN];
  __shared__ float sScale[CIN], sShift[CIN];
  const int t = threadIdx.x, wv = t >> 5, lane = t & 31;
  for (int i = t; i < CO * CIN; i += 256) sW[i] = (h16)W[i];
  for (int i = t; i < CIN; i += 256) { sScale[i] = scale[i]; sShift[i] = shift[i]; }
  __syncthreads();
  const int ntiles = R >> 4;
  const int wstride = gridDim.x * 8;
  h16* Aw = sA[wv];
  constexpr int HALF = CIN >> 1;                 // halves per lane (2 lanes per row)
  for (int tile = blockIdx.x * 8 + wv; tile < ntiles; tile += wstride) {
    const int r0 = tile << 4;
    {
      const int row = lane >> 1, c0 = (lane & 1) * HALF;
      const h16* src = A + (size_t)(r0 + row) * CIN + c0;
      h16* dst = Aw + row * CIN + c0;
#pragma unroll
      for (int c = 0; c < HALF; c += 8) {        // 128-bit loads, BN+ReLU, 128-bit LDS stores
        uint4 v = *(const uint4*)(src + c);
        h16 tmp[8]; *(uint4*)tmp = v;
#pragma unroll
        for (int e = 0; e < 8; ++e) {
          const int ch = c0 + c + e;
          float f = (float)tmp[e];
          f = fmaxf(0.f, f * sScale[ch] + sShift[ch]);
          tmp[e] = (h16)f;
        }
        *(uint4*)(dst + c) = *(uint4*)tmp;
      }
      if (tile + wstride < ntiles)               // global_prefetch_b8 for next tile
        __builtin_prefetch(A + (size_t)(r0 + (wstride << 4) + row) * CIN + c0, 0, 1);
    }
    asm volatile("s_wait_dscnt 0" ::: "memory");
    v16h a[CIN >> 5];
    const int m = lane & 15, nn = lane & 15;
#pragma unroll
    for (int kt = 0; kt < (CIN >> 5); ++kt)
#pragma unroll
      for (int h = 0; h < 16; ++h) a[kt][h] = Aw[m * CIN + kt * 32 + frag_k(lane, h)];
#pragma unroll
    for (int ct = 0; ct < (CO >> 4); ++ct) {
      v8f acc = {};
#pragma unroll
      for (int kt = 0; kt < (CIN >> 5); ++kt) {
        v16h bf;
#pragma unroll
        for (int h = 0; h < 16; ++h)
          bf[h] = sW[(ct * 16 + nn) * CIN + kt * 32 + frag_k(lane, h)];
        acc = __builtin_amdgcn_wmma_f32_16x16x32_f16(false, a[kt], false, bf, (short)0, acc, false, false);
      }
      epilogue(acc, bias, X, sum, sumsq, r0, ct, CO, lane);
    }
  }
}

// =====================  4. BN finalize (per-channel scale/shift)  =====================
__global__ void bn_finalize_kernel(const float* __restrict__ sum, const float* __restrict__ sumsq,
                                   const float* __restrict__ g, const float* __restrict__ be,
                                   float* __restrict__ scale, float* __restrict__ shift,
                                   int co, float invR) {
  const int o = threadIdx.x;
  if (o < co) {
    const float mean = sum[o] * invR;
    const float var  = sumsq[o] * invR - mean * mean;
    const float inv  = rsqrtf(var + BN_EPS);
    const float sc   = g[o] * inv;
    scale[o] = sc;
    shift[o] = be[o] - mean * sc;
  }
}

// =====================  5. Final BN+ReLU + max over K, transposed store =====================
__global__ void maxpool_kernel(const h16* __restrict__ X, const float* __restrict__ scale,
                               const float* __restrict__ shift, float* __restrict__ out,
                               int K, int co, int obase) {
  const int q = blockIdx.x;                      // b*S + s
  const int o = threadIdx.x;
  if (o >= co) return;
  const int b = q / S_, s = q % S_;
  const float sc = scale[o], sh = shift[o];
  float m = 0.f;                                 // relu output is >= 0
  const h16* Xp = X + (size_t)q * K * co + o;
  for (int kk = 0; kk < K; ++kk) {
    float v = (float)Xp[(size_t)kk * co] * sc + sh;
    m = fmaxf(m, fmaxf(v, 0.f));
  }
  out[(size_t)B_ * 3 * S_ + (size_t)b * 320 * S_ + (size_t)(obase + o) * S_ + s] = m;
}

// =====================  host orchestration  =====================
extern "C" void kernel_launch(void* const* d_in, const int* in_sizes, int n_in,
                              void* d_out, int out_size, void* d_ws, size_t ws_size,
                              hipStream_t stream) {
  const float* xyz = (const float*)d_in[0];
  const float* pts = (const float*)d_in[1];
  auto P = [&](int br, int ly, int which) -> const float* {
    return (const float*)d_in[2 + br * 12 + ly * 4 + which];  // 0=W 1=b 2=g 3=be
  };

  char*  ws      = (char*)d_ws;
  int*   fpsidx  = (int*)ws;                       // 32 KB
  float* nxyz    = (float*)(ws + 32768);           // (B,S,3)
  int*   gidx[3] = { (int*)(ws + 131072), (int*)(ws + 655360), (int*)(ws + 1703936) };
  float* stats   = (float*)(ws + 6 * 1024 * 1024); // 9 slots x 512 floats
  h16*   bufA    = (h16*)(ws + 8ull * 1024 * 1024);
  h16*   bufB    = (h16*)(ws + 8ull * 1024 * 1024 + 268435456ull);
  float* out     = (float*)d_out;

  fps_kernel<<<B_, 256, 0, stream>>>(xyz, fpsidx, nxyz, out);

  const int   Ks[3]  = {16, 32, 128};
  const float r2[3]  = {0.01f, 0.04f, 0.16f};
  const int   ob[3]  = {0, 64, 192};

  for (int br = 0; br < 3; ++br) {
    const int K = Ks[br], R = B_ * S_ * K;
    const float invR = 1.f / (float)R;
    ballquery_kernel<<<(B_ * S_) / 8, 256, 0, stream>>>(xyz, nxyz, gidx[br], r2[br], K);

    float* st0 = stats + (br * 3 + 0) * 512;   // [sum|sumsq|scale|shift] x128
    float* st1 = stats + (br * 3 + 1) * 512;
    float* st2 = stats + (br * 3 + 2) * 512;
    hipMemsetAsync(st0, 0, 1024, stream);
    hipMemsetAsync(st1, 0, 1024, stream);
    hipMemsetAsync(st2, 0, 1024, stream);

    if (br == 0) {
      gemm_l1_kernel<32><<<512, 256, 0, stream>>>(xyz, pts, nxyz, gidx[br], P(0,0,0), P(0,0,1),
                                                  bufA, st0, st0 + 128, K);
      bn_finalize_kernel<<<1, 128, 0, stream>>>(st0, st0 + 128, P(0,0,2), P(0,0,3),
                                                st0 + 256, st0 + 384, 32, invR);
      gemm_bn_kernel<32, 32><<<512, 256, 0, stream>>>(bufA, st0 + 256, st0 + 384, P(0,1,0), P(0,1,1),
                                                      bufB, st1, st1 + 128, R);
      bn_finalize_kernel<<<1, 128, 0, stream>>>(st1, st1 + 128, P(0,1,2), P(0,1,3),
                                                st1 + 256, st1 + 384, 32, invR);
      gemm_bn_kernel<32, 64><<<512, 256, 0, stream>>>(bufB, st1 + 256, st1 + 384, P(0,2,0), P(0,2,1),
                                                      bufA, st2, st2 + 128, R);
      bn_finalize_kernel<<<1, 128, 0, stream>>>(st2, st2 + 128, P(0,2,2), P(0,2,3),
                                                st2 + 256, st2 + 384, 64, invR);
      maxpool_kernel<<<B_ * S_, 64, 0, stream>>>(bufA, st2 + 256, st2 + 384, out, K, 64, ob[br]);
    } else if (br == 1) {
      gemm_l1_kernel<64><<<512, 256, 0, stream>>>(xyz, pts, nxyz, gidx[br], P(1,0,0), P(1,0,1),
                                                  bufA, st0, st0 + 128, K);
      bn_finalize_kernel<<<1, 128, 0, stream>>>(st0, st0 + 128, P(1,0,2), P(1,0,3),
                                                st0 + 256, st0 + 384, 64, invR);
      gemm_bn_kernel<64, 64><<<512, 256, 0, stream>>>(bufA, st0 + 256, st0 + 384, P(1,1,0), P(1,1,1),
                                                      bufB, st1, st1 + 128, R);
      bn_finalize_kernel<<<1, 128, 0, stream>>>(st1, st1 + 128, P(1,1,2), P(1,1,3),
                                                st1 + 256, st1 + 384, 64, invR);
      gemm_bn_kernel<64, 128><<<512, 256, 0, stream>>>(bufB, st1 + 256, st1 + 384, P(1,2,0), P(1,2,1),
                                                       bufA, st2, st2 + 128, R);
      bn_finalize_kernel<<<1, 128, 0, stream>>>(st2, st2 + 128, P(1,2,2), P(1,2,3),
                                                st2 + 256, st2 + 384, 128, invR);
      maxpool_kernel<<<B_ * S_, 128, 0, stream>>>(bufA, st2 + 256, st2 + 384, out, K, 128, ob[br]);
    } else {
      gemm_l1_kernel<64><<<512, 256, 0, stream>>>(xyz, pts, nxyz, gidx[br], P(2,0,0), P(2,0,1),
                                                  bufA, st0, st0 + 128, K);
      bn_finalize_kernel<<<1, 128, 0, stream>>>(st0, st0 + 128, P(2,0,2), P(2,0,3),
                                                st0 + 256, st0 + 384, 64, invR);
      gemm_bn_kernel<64, 96><<<512, 256, 0, stream>>>(bufA, st0 + 256, st0 + 384, P(2,1,0), P(2,1,1),
                                                      bufB, st1, st1 + 128, R);
      bn_finalize_kernel<<<1, 128, 0, stream>>>(st1, st1 + 128, P(2,1,2), P(2,1,3),
                                                st1 + 256, st1 + 384, 96, invR);
      gemm_bn_kernel<96, 128><<<512, 256, 0, stream>>>(bufB, st1 + 256, st1 + 384, P(2,2,0), P(2,2,1),
                                                       bufA, st2, st2 + 128, R);
      bn_finalize_kernel<<<1, 128, 0, stream>>>(st2, st2 + 128, P(2,2,2), P(2,2,3),
                                                st2 + 256, st2 + 384, 128, invR);
      maxpool_kernel<<<B_ * S_, 128, 0, stream>>>(bufA, st2 + 256, st2 + 384, out, K, 128, ob[br]);
    }
  }
}
